// TemporalAttention_15556371546852
// MI455X (gfx1250) — matrix operand order repaired and checked by reading, not compiled, not run on hardware
//
#include <hip/hip_runtime.h>
#include <hip/hip_bf16.h>

// B=2, M=16, C=256, H=W=64, HW=4096, T=1024
typedef __attribute__((ext_vector_type(16))) _Float16 v16h;
typedef __attribute__((ext_vector_type(8)))  float    v8f;
typedef __attribute__((ext_vector_type(4)))  float    v4f;

#define TA_C   256
#define TA_M   16
#define TA_HW  4096
#define TA_T   1024
#define TA_PIX 16          // pixels per block (x-tile fits LDS as f16)
#define TA_PAD 16          // f16 pad per wgt LDS row (bank spread)

// ---------------------------------------------------------------------------
// Prep: q = Wq@t + bq ; qk = scale * Wk^T q ; sb = scale * q.bk ; Wv -> f16
// ---------------------------------------------------------------------------
__global__ __launch_bounds__(256) void ta_prep(
    const float* __restrict__ tr, const float* __restrict__ Wq,
    const float* __restrict__ bq, const float* __restrict__ Wk,
    const float* __restrict__ bk, const float* __restrict__ Wv,
    float* __restrict__ qks, float* __restrict__ sbs,
    _Float16* __restrict__ wvh)
{
    __shared__ float qs[2][TA_C];
    __shared__ float red[2][TA_C];
    const int c = threadIdx.x;
    const float scale = 0.0625f;               // C^-0.5 = 1/16

    // q[b][c] = bq[c] + dot(Wq[c,:], t[b,:])
    float q0 = bq[c], q1 = bq[c];
    const float* wrow = Wq + (size_t)c * TA_T;
    for (int t = 0; t < TA_T; ++t) {
        const float w = wrow[t];
        q0 += w * tr[t];
        q1 += w * tr[TA_T + t];
    }
    qs[0][c] = q0;  qs[1][c] = q1;
    red[0][c] = q0 * bk[c];
    red[1][c] = q1 * bk[c];
    __syncthreads();

    // qk[b][c] = scale * sum_o Wk[o][c] * q[b][o]   (coalesced over c)
    float a0 = 0.f, a1 = 0.f;
    for (int o = 0; o < TA_C; ++o) {
        const float w = Wk[(size_t)o * TA_C + c];
        a0 += w * qs[0][o];
        a1 += w * qs[1][o];
    }
    qks[c]        = a0 * scale;
    qks[TA_C + c] = a1 * scale;

    if (c < 2) {                                // sb[b] = scale * q.bk
        float s = 0.f;
        for (int i = 0; i < TA_C; ++i) s += red[c][i];
        sbs[c] = s * scale;
    }
    // Wv -> f16 (128 KB, L2-resident afterwards)
    for (int i = c; i < TA_C * TA_C; i += 256) wvh[i] = (_Float16)Wv[i];
}

// ---------------------------------------------------------------------------
// Fused attention. One block = 16 contiguous pixels of one batch element.
// Single global pass over x: phase 1 streams x (non-temporal), scores in f32,
// caches the tile as f16 in LDS; phases 2-3 run entirely out of LDS.
// ---------------------------------------------------------------------------
__global__ __launch_bounds__(256) void ta_main(
    const float* __restrict__ x,          // [B,M,C,HW]
    const float* __restrict__ qks,        // [B,C]  (pre-scaled)
    const float* __restrict__ sbs,        // [B]    (pre-scaled)
    const _Float16* __restrict__ wvh,     // [C,C]  f16
    const float* __restrict__ bv,         // [C]
    float* __restrict__ out)              // [B,C,HW]
{
    __shared__ alignas(16) _Float16 xt[TA_M][TA_C][TA_PIX];       // 128 KB
    __shared__ float s_part[16][TA_M][TA_PIX];                    //  16 KB
    __shared__ float sc[TA_M][TA_PIX];                            //   1 KB
    __shared__ alignas(16) _Float16 wgt[TA_PIX][TA_C + TA_PAD];   // 8.5 KB

    const int tid = threadIdx.x;
    const int p   = tid & 15;           // pixel within tile
    const int cg  = tid >> 4;           // channel group (16 ch each)
    const int blk = blockIdx.x;         // 0..511
    const int b   = blk >> 8;
    const int hw0 = (blk & 255) * TA_PIX;

    const float* xb = x + (size_t)b * TA_M * TA_C * TA_HW;
    const size_t mstride = (size_t)TA_C * TA_HW;

    // ---- phase 1: stream x once (NT), score partials + f16 LDS tile -------
    float s[TA_M];
    #pragma unroll
    for (int m = 0; m < TA_M; ++m) s[m] = 0.f;
    const float* qk = qks + (size_t)b * TA_C;
    for (int cc = 0; cc < 16; ++cc) {
        const int c = cg * 16 + cc;
        const float w = qk[c];                                    // scalar
        const float* px = xb + (size_t)c * TA_HW + hw0 + p;
        #pragma unroll
        for (int m = 0; m < TA_M; ++m) {
            const float v = __builtin_nontemporal_load(px + m * mstride);
            s[m] += w * v;
            xt[m][c][p] = (_Float16)v;
        }
    }
    #pragma unroll
    for (int m = 0; m < TA_M; ++m) s_part[cg][m][p] = s[m];
    __syncthreads();

    // reduce the 16 channel-group partials + bias term (one thread each)
    {
        const int m = tid >> 4, pp = tid & 15;
        float acc = sbs[b];
        #pragma unroll
        for (int g = 0; g < 16; ++g) acc += s_part[g][m][pp];
        sc[m][pp] = acc;
    }
    __syncthreads();

    // ---- softmax over M (one thread per pixel) ----------------------------
    if (tid < TA_PIX) {
        float mx = sc[0][tid];
        #pragma unroll
        for (int m = 1; m < TA_M; ++m) mx = fmaxf(mx, sc[m][tid]);
        float e[TA_M], sum = 0.f;
        #pragma unroll
        for (int m = 0; m < TA_M; ++m) { e[m] = __expf(sc[m][tid] - mx); sum += e[m]; }
        const float inv = 1.f / sum;
        #pragma unroll
        for (int m = 0; m < TA_M; ++m) sc[m][tid] = e[m] * inv;
    }
    __syncthreads();

    // ---- phase 2: weighted value map from LDS tile -> wgt[p][c] f16 -------
    float pr[TA_M];
    #pragma unroll
    for (int m = 0; m < TA_M; ++m) pr[m] = sc[m][p];
    for (int cc = 0; cc < 16; ++cc) {
        const int c = cg * 16 + cc;
        float acc = 0.f;
        #pragma unroll
        for (int m = 0; m < TA_M; ++m)
            acc += pr[m] * (float)xt[m][c][p];
        wgt[p][c] = (_Float16)acc;
    }
    __syncthreads();

    // ---- phase 3: out_tile = Wv @ wgt^T + bv  (v_wmma f16->f32) -----------
    const int wave  = tid >> 5;
    const int lane  = tid & 31;
    const int lhalf = lane >> 4;        // 0: lanes 0-15, 1: lanes 16-31
    const int l16   = lane & 15;

    #pragma unroll
    for (int i = 0; i < 2; ++i) {
        const int o_t = wave + i * 8;   // 16 output-channel tiles / 8 waves
        v8f acc = {};
        for (int k0 = 0; k0 < TA_C; k0 += 32) {
            // A (16x32 f16): lane half 0 holds K {0-7,16-23}, half 1 {8-15,24-31}
            union { v16h v; v4f f[2]; } a, bf;
            const _Float16* arow = wvh + (size_t)(o_t * 16 + l16) * TA_C
                                       + k0 + lhalf * 8;
            a.f[0] = *(const v4f*)(arow);
            a.f[1] = *(const v4f*)(arow + 16);
            // B (32x16 f16): lane half 0 holds K {0-7,16-23}, half 1 {8-15,24-31}
            const _Float16* brow = &wgt[l16][k0 + lhalf * 8];
            bf.f[0] = *(const v4f*)(brow);
            bf.f[1] = *(const v4f*)(brow + 16);
            acc = __builtin_amdgcn_wmma_f32_16x16x32_f16(
                      false, a.v, false, bf.v, (short)0, acc, false, false);
        }
        // D 16x16 f32: VGPR r -> M = r + 8*lhalf, N = l16
        #pragma unroll
        for (int r = 0; r < 8; ++r) {
            const int o = o_t * 16 + r + lhalf * 8;
            out[((size_t)b * TA_C + o) * TA_HW + hw0 + l16] = acc[r] + bv[o];
        }
    }
}

// ---------------------------------------------------------------------------
extern "C" void kernel_launch(void* const* d_in, const int* in_sizes, int n_in,
                              void* d_out, int out_size, void* d_ws, size_t ws_size,
                              hipStream_t stream) {
    (void)in_sizes; (void)n_in; (void)out_size; (void)ws_size;
    const float* tr = (const float*)d_in[0];
    const float* x  = (const float*)d_in[1];
    const float* Wq = (const float*)d_in[2];
    const float* bq = (const float*)d_in[3];
    const float* Wk = (const float*)d_in[4];
    const float* bk = (const float*)d_in[5];
    const float* Wv = (const float*)d_in[6];
    const float* bv = (const float*)d_in[7];
    float* out = (float*)d_out;

    char* ws = (char*)d_ws;
    float*    qks = (float*)ws;                 // 512 f32
    float*    sbs = (float*)(ws + 512 * 4);     //   2 f32
    _Float16* wvh = (_Float16*)(ws + 4096);     // 65536 f16 (128 KB)

    ta_prep<<<1, 256, 0, stream>>>(tr, Wq, bq, Wk, bk, Wv, qks, sbs, wvh);
    ta_main<<<512, 256, 0, stream>>>(x, qks, sbs, wvh, bv, out);
}
